// TimeMixer_49787260895836
// MI455X (gfx1250) — matrix operand order repaired and checked by reading, not compiled, not run on hardware
//
#include <hip/hip_runtime.h>
#include <hip/hip_bf16.h>
#include <math.h>

// ---------------------------------------------------------------------------
// Types for WMMA (CDNA5 / gfx1250, wave32)
// ---------------------------------------------------------------------------
typedef __bf16          v16bf  __attribute__((ext_vector_type(16)));
typedef float           v8f    __attribute__((ext_vector_type(8)));
typedef unsigned short  u16x16 __attribute__((ext_vector_type(16)));
typedef unsigned short  u16x8  __attribute__((ext_vector_type(8)));

__device__ __forceinline__ unsigned short f2bf(float f) {
    unsigned int u = __float_as_uint(f);
    unsigned int r = u + 0x7FFFu + ((u >> 16) & 1u);   // round-to-nearest-even
    return (unsigned short)(r >> 16);
}
__device__ __forceinline__ float sigm(float z) { return 1.0f / (1.0f + expf(-z)); }

// ---------------------------------------------------------------------------
// Tiled bf16 WMMA GEMM:  C(f32, MxN) = A(bf16, MxK) * Bt(bf16, NxK)^T
//   A  : M x K row-major (lda), M % 128 == 0, K % 32 == 0, lda % 8 == 0
//   Bt : N x K row-major (ldb) -- i.e. B stored column-major ("weight layout")
// Both LDS tiles are K-contiguous, so every WMMA fragment is two contiguous
// 16-byte runs -> pairs of ds_load_b128 (no v_mov_b16 repacking).
// 8 waves: 4 along M (32 rows), 2 along N (64 cols); 2x4 accumulators/wave.
// ---------------------------------------------------------------------------
#define BM 128
#define BN 128
#define BK 32

__global__ __launch_bounds__(256) void gemm_bf16_kernel(
    const unsigned short* __restrict__ A, const unsigned short* __restrict__ Bt,
    float* __restrict__ C, int M, int N, int K, int lda, int ldb, int ldc)
{
    __shared__ unsigned short As[BM * BK];   // [row][k]
    __shared__ unsigned short Bs[BN * BK];   // [col][k]

    const int tid  = threadIdx.x;
    const int lane = tid & 31;
    const int wid  = tid >> 5;
    const int wm   = wid & 3;    // wave row  (0..3) -> 32 rows
    const int wn   = wid >> 2;   // wave col  (0..1) -> 64 cols
    const int bm   = blockIdx.x * BM;
    const int bn   = blockIdx.y * BN;

    v8f zero = {0.f, 0.f, 0.f, 0.f, 0.f, 0.f, 0.f, 0.f};
    v8f acc[2][4];
    #pragma unroll
    for (int i = 0; i < 2; i++)
        #pragma unroll
        for (int j = 0; j < 4; j++) acc[i][j] = zero;

    const int srow = tid >> 1;           // 128 rows/cols, 2 threads each
    const int scol = (tid & 1) * 16;     // 16 bf16 per thread

    const u16x8 zv = {0, 0, 0, 0, 0, 0, 0, 0};

    for (int k0 = 0; k0 < K; k0 += BK) {
        // ---- stage A tile (always in-range: M%128==0, K%32==0) ----
        {
            const u16x8* src = reinterpret_cast<const u16x8*>(
                &A[(size_t)(bm + srow) * lda + k0 + scol]);
            u16x8* dst = reinterpret_cast<u16x8*>(&As[srow * BK + scol]);
            dst[0] = src[0];
            dst[1] = src[1];
        }
        // ---- stage B tile from N x K weight layout (row guard on N) ----
        {
            u16x8* dst = reinterpret_cast<u16x8*>(&Bs[srow * BK + scol]);
            if (bn + srow < N) {
                const u16x8* src = reinterpret_cast<const u16x8*>(
                    &Bt[(size_t)(bn + srow) * ldb + k0 + scol]);
                dst[0] = src[0];
                dst[1] = src[1];
            } else {
                dst[0] = zv;
                dst[1] = zv;
            }
        }
        // prefetch next K slab into L2 (global_prefetch_b8)
        if (k0 + BK < K) {
            __builtin_prefetch(&A[(size_t)(bm + srow) * lda + k0 + BK + scol], 0, 1);
            if (bn + srow < N)
                __builtin_prefetch(&Bt[(size_t)(bn + srow) * ldb + k0 + BK + scol], 0, 1);
        }
        __syncthreads();

        // ---- build fragments: contiguous k order, k = 16*(lane/16) + e, same for A and B ----
        const int lr  = lane & 15;
        const int akb = (lane < 16) ? 0 : 16;   // A: K-base per half-wave
        const int bkb = (lane < 16) ? 0 : 16;   // B: K-base per half-wave
        v16bf af[2], bfr[4];
        #pragma unroll
        for (int mi = 0; mi < 2; mi++) {
            const int row = wm * 32 + mi * 16 + lr;
            u16x16 tmp;
            #pragma unroll
            for (int e = 0; e < 16; e++) {
                const int kk = akb + e;
                tmp[e] = As[row * BK + kk];
            }
            af[mi] = __builtin_bit_cast(v16bf, tmp);
        }
        #pragma unroll
        for (int ni = 0; ni < 4; ni++) {
            const int col = wn * 64 + ni * 16 + lr;
            u16x16 tmp;
            #pragma unroll
            for (int e = 0; e < 16; e++)
                tmp[e] = Bs[col * BK + bkb + e];    // 16 contiguous bf16
            bfr[ni] = __builtin_bit_cast(v16bf, tmp);
        }
        #pragma unroll
        for (int mi = 0; mi < 2; mi++)
            #pragma unroll
            for (int ni = 0; ni < 4; ni++)
                acc[mi][ni] = __builtin_amdgcn_wmma_f32_16x16x32_bf16(
                    false, af[mi], false, bfr[ni], (short)0, acc[mi][ni],
                    false, false);
        __syncthreads();
    }

    // ---- store: C/D layout -> lane = col, VGPR r = row (+8 for upper lanes)
    const int lr     = lane & 15;
    const int rowoff = (lane < 16) ? 0 : 8;
    #pragma unroll
    for (int mi = 0; mi < 2; mi++) {
        #pragma unroll
        for (int ni = 0; ni < 4; ni++) {
            const int gcol = bn + wn * 64 + ni * 16 + lr;
            if (gcol >= N) continue;
            const int grow0 = bm + wm * 32 + mi * 16 + rowoff;
            #pragma unroll
            for (int rr = 0; rr < 8; rr++)
                C[(size_t)(grow0 + rr) * ldc + gcol] = acc[mi][ni][rr];
        }
    }
}

// ---------------------------------------------------------------------------
// Weight prep kernels
// ---------------------------------------------------------------------------
// plain f32 -> bf16 convert (weights already in N x K layout)
__global__ void cvt_kernel(const float* __restrict__ src,
                           unsigned short* __restrict__ dst, int count)
{
    const int i = blockIdx.x * 256 + threadIdx.x;
    if (i >= count) return;
    dst[i] = f2bf(src[i]);
}

// src: K x N (f32) -> dst: N x Kpad (bf16), zero-pad cols K..Kpad-1
__global__ void transpose_pad_kernel(const float* __restrict__ src,
                                     unsigned short* __restrict__ dst,
                                     int K, int Kpad, int N)
{
    const int i = blockIdx.x * 256 + threadIdx.x;
    if (i >= N * Kpad) return;
    const int n = i / Kpad, k = i % Kpad;
    dst[i] = (k < K) ? f2bf(src[(size_t)k * N + n]) : (unsigned short)0;
}

// src: M x K (f32) -> dst: M x Kpad (bf16), optional tanh, zero-pad cols
__global__ void act_cvt_kernel(const float* __restrict__ src,
                               unsigned short* __restrict__ dst,
                               int M, int K, int Kpad, int do_tanh)
{
    const int i = blockIdx.x * 256 + threadIdx.x;
    if (i >= M * Kpad) return;
    const int m = i / Kpad, k = i % Kpad;
    float v = (k < K) ? src[(size_t)m * K + k] : 0.0f;
    if (do_tanh) v = tanhf(v);
    dst[i] = f2bf(v);
}

// ---------------------------------------------------------------------------
// Token-shift: dx = xprev - x ; base = x + dx * time_maa_x  (bf16)
// ---------------------------------------------------------------------------
__global__ __launch_bounds__(256) void prep_kernel(
    const float* __restrict__ x, const float* __restrict__ maa_x,
    float* __restrict__ dx, unsigned short* __restrict__ basebf,
    int T, int C)
{
    const int n = blockIdx.x;
    const int t = n % T;
    #pragma unroll
    for (int q = 0; q < 4; q++) {
        const int c = threadIdx.x * 4 + q;
        const size_t idx = (size_t)n * C + c;
        const float xv = x[idx];
        const float xp = (t == 0) ? 0.0f : x[idx - C];
        const float d  = xp - xv;
        dx[idx]     = d;
        basebf[idx] = f2bf(xv + d * maa_x[c]);
    }
}

// xmix_i = x + dx * (time_maa_i + m_i)   -> bf16
__global__ __launch_bounds__(256) void mix_kernel(
    const float* __restrict__ x, const float* __restrict__ dx,
    const float* __restrict__ maa_row, const float* __restrict__ m,
    unsigned short* __restrict__ out, int C)
{
    const int n = blockIdx.x;
    #pragma unroll
    for (int q = 0; q < 4; q++) {
        const int c = threadIdx.x * 4 + q;
        const size_t idx = (size_t)n * C + c;
        out[idx] = f2bf(x[idx] + dx[idx] * (maa_row[c] + m[idx]));
    }
}

// ---------------------------------------------------------------------------
// expand_loras: w/a/ma/mk, per-head L2 normalize kk, k modulation, b = kk*a
// Per-head reduction via 16-lane shuffle groups (wave32: 2 heads per wave).
// In-place: kbuf <- k_mod, d2 <- wdec = exp(-exp(w)).
// ---------------------------------------------------------------------------
__global__ __launch_bounds__(256) void expand_kernel(
    const float* __restrict__ td, const float* __restrict__ aaaaa,
    const float* __restrict__ misc_a, const float* __restrict__ misc_k,
    float* __restrict__ kbuf, const float* __restrict__ kk2,
    float* __restrict__ d2, const float* __restrict__ a2,
    const float* __restrict__ ma2, const float* __restrict__ mk2,
    float* __restrict__ kkn, float* __restrict__ bb, int C)
{
    const int n = blockIdx.x;
    const int tid = threadIdx.x;
    float kraw[4], kkv[4], wv[4], av[4], mav[4], mkv[4];
    float ss = 0.0f;
    #pragma unroll
    for (int q = 0; q < 4; q++) {
        const int c = tid * 4 + q;
        const size_t idx = (size_t)n * C + c;
        const float kv  = kbuf[idx];
        const float kkr = kv + kk2[idx];
        kraw[q] = kv;
        kkv[q]  = kkr;
        ss     += kkr * kkr;
        const float u = -(td[c] + d2[idx]);                 // softplus(-z)
        const float sp = fmaxf(u, 0.0f) + log1pf(expf(-fabsf(u)));
        wv[q]  = -sp - 0.5f;
        av[q]  = sigm(aaaaa[c]  + a2[idx]);
        mav[q] = sigm(misc_a[c] + ma2[idx]);
        mkv[q] = sigm(misc_k[c] + mk2[idx]);
    }
    #pragma unroll
    for (int off = 8; off >= 1; off >>= 1) ss += __shfl_xor(ss, off, 16);
    const float inv = 1.0f / fmaxf(sqrtf(ss), 1e-12f);
    #pragma unroll
    for (int q = 0; q < 4; q++) {
        const int c = tid * 4 + q;
        const size_t idx = (size_t)n * C + c;
        const float kkn_v = kkv[q] * inv;
        kkn[idx] = kkn_v;
        bb[idx]  = kkn_v * av[q];
        float km = kraw[q] * (mav[q] + av[q] * (1.0f - mav[q]));
        km *= expf(fminf(wv[q] * mkv[q], 0.0f));
        kbuf[idx] = km;
        d2[idx]   = expf(-expf(wv[q]));                     // wdec
    }
}

// ---------------------------------------------------------------------------
// Per-head selective-decay linear-attention scan.
// One block per (b,h): 64 threads = 2 wave32s. Thread i owns row i of the
// 64x64 state S (64 VGPRs). Shared vectors staged through LDS each step.
// ---------------------------------------------------------------------------
__global__ __launch_bounds__(64) void scan_kernel(
    const float* __restrict__ wdec, const float* __restrict__ r,
    const float* __restrict__ k,    const float* __restrict__ kkn,
    const float* __restrict__ b,    const float* __restrict__ v,
    float* __restrict__ y, int B, int T, int H, int C)
{
    const int bh = blockIdx.x;
    const int bb_ = bh / H, h = bh % H;
    const int i = threadIdx.x;
    float S[64];
    #pragma unroll
    for (int j = 0; j < 64; j++) S[j] = 0.0f;

    __shared__ float sw[64], sr[64], sk[64], sa[64], sb[64];
    for (int t = 0; t < T; t++) {
        const size_t off = ((size_t)(bb_ * T + t)) * C + h * 64;
        sw[i] = wdec[off + i];
        sr[i] = r[off + i];
        sk[i] = k[off + i];
        sa[i] = -kkn[off + i];
        sb[i] = b[off + i];
        const float vi = v[off + i];
        __syncthreads();
        float Sa = 0.0f;
        #pragma unroll
        for (int j = 0; j < 64; j++) Sa += S[j] * sa[j];
        float yi = 0.0f;
        #pragma unroll
        for (int j = 0; j < 64; j++) {
            const float s = S[j] * sw[j] + Sa * sb[j] + vi * sk[j];
            S[j] = s;
            yi += s * sr[j];
        }
        y[off + i] = yi;
        __syncthreads();
    }
}

// ---------------------------------------------------------------------------
// Fused per-head GroupNorm + first-token bonus + gate -> bf16 for final GEMM
// ---------------------------------------------------------------------------
__global__ __launch_bounds__(256) void gnorm_kernel(
    const float* __restrict__ y, const float* __restrict__ r,
    const float* __restrict__ k, const float* __restrict__ v,
    const float* __restrict__ g, const float* __restrict__ ln_w,
    const float* __restrict__ ln_b, const float* __restrict__ faaaa,
    unsigned short* __restrict__ ybf, int C)
{
    const int n = blockIdx.x;
    const int tid = threadIdx.x;
    float yv[4];
    float s1 = 0.0f, s2 = 0.0f, sbn = 0.0f;
    #pragma unroll
    for (int q = 0; q < 4; q++) {
        const int c = tid * 4 + q;
        const size_t idx = (size_t)n * C + c;
        yv[q] = y[idx];
        s1 += yv[q];
        s2 += yv[q] * yv[q];
        sbn += r[idx] * k[idx] * faaaa[c];
    }
    #pragma unroll
    for (int off = 8; off >= 1; off >>= 1) {
        s1  += __shfl_xor(s1, off, 16);
        s2  += __shfl_xor(s2, off, 16);
        sbn += __shfl_xor(sbn, off, 16);
    }
    const float mu  = s1 * (1.0f / 64.0f);
    const float var = s2 * (1.0f / 64.0f) - mu * mu;
    const float rs  = rsqrtf(var + 64e-5f);
    #pragma unroll
    for (int q = 0; q < 4; q++) {
        const int c = tid * 4 + q;
        const size_t idx = (size_t)n * C + c;
        const float yn = (yv[q] - mu) * rs * ln_w[c] + ln_b[c];
        const float o  = (yn + sbn * v[idx]) * g[idx];
        ybf[idx] = f2bf(o);
    }
}

// ---------------------------------------------------------------------------
// Host orchestration
// ---------------------------------------------------------------------------
extern "C" void kernel_launch(void* const* d_in, const int* in_sizes, int n_in,
                              void* d_out, int out_size, void* d_ws, size_t ws_size,
                              hipStream_t stream)
{
    (void)n_in; (void)out_size; (void)ws_size; (void)in_sizes;
    const int Bb = 4, T = 1024, C = 1024, H = 16;
    const int BT = Bb * T;

    const float* x          = (const float*)d_in[0];
    const float* time_maa_x = (const float*)d_in[2];
    const float* time_maa   = (const float*)d_in[3];
    const float* maa_w1     = (const float*)d_in[4];   // (128, C)
    const float* maa_w2     = (const float*)d_in[5];   // (4, C, 32)
    const float* time_decay = (const float*)d_in[6];
    const float* dec_w1     = (const float*)d_in[7];   // (64, C)
    const float* dec_w2     = (const float*)d_in[8];   // (64, C)
    const float* time_aaaaa = (const float*)d_in[9];
    const float* aaa_w1     = (const float*)d_in[10];  // (16, C)
    const float* aaa_w2     = (const float*)d_in[11];  // (16, C)
    const float* kkk_w1     = (const float*)d_in[12];
    const float* kkk_w2     = (const float*)d_in[13];
    const float* gate_w1    = (const float*)d_in[14];  // (128, C)
    const float* gate_w2    = (const float*)d_in[15];  // (128, C)
    const float* time_misc_a= (const float*)d_in[16];
    const float* ma_w1      = (const float*)d_in[17];
    const float* ma_w2      = (const float*)d_in[18];
    const float* time_misc_k= (const float*)d_in[19];
    const float* mk_w1      = (const float*)d_in[20];
    const float* mk_w2      = (const float*)d_in[21];
    const float* time_faaaa = (const float*)d_in[22];
    const float* Wr         = (const float*)d_in[23];
    const float* Wk         = (const float*)d_in[24];
    const float* Wv         = (const float*)d_in[25];
    const float* Wo         = (const float*)d_in[26];
    const float* ln_w       = (const float*)d_in[27];
    const float* ln_b       = (const float*)d_in[28];

    char* ws = (char*)d_ws;
    size_t off = 0;
    auto alloc = [&](size_t bytes) -> char* {
        char* p = ws + off;
        off += (bytes + 255) & ~(size_t)255;
        return p;
    };

    // --- bf16 weights, all in N x K ("weight") layout for the GEMM B side ---
    unsigned short* wWr   = (unsigned short*)alloc((size_t)C * C * 2);
    unsigned short* wWk   = (unsigned short*)alloc((size_t)C * C * 2);
    unsigned short* wWv   = (unsigned short*)alloc((size_t)C * C * 2);
    unsigned short* wWo   = (unsigned short*)alloc((size_t)C * C * 2);
    unsigned short* wMaa1 = (unsigned short*)alloc((size_t)128 * C * 2);
    unsigned short* wMaa2 = (unsigned short*)alloc((size_t)4 * C * 32 * 2);
    unsigned short* wGate1= (unsigned short*)alloc((size_t)128 * C * 2);
    unsigned short* wGate2T=(unsigned short*)alloc((size_t)C * 128 * 2);
    unsigned short* wDec1 = (unsigned short*)alloc((size_t)64 * C * 2);
    unsigned short* wDec2T= (unsigned short*)alloc((size_t)C * 64 * 2);
    unsigned short* wA1   = (unsigned short*)alloc((size_t)16 * C * 2);
    unsigned short* wK1   = (unsigned short*)alloc((size_t)16 * C * 2);
    unsigned short* wMa1  = (unsigned short*)alloc((size_t)16 * C * 2);
    unsigned short* wMk1  = (unsigned short*)alloc((size_t)16 * C * 2);
    unsigned short* wA2T  = (unsigned short*)alloc((size_t)C * 32 * 2);
    unsigned short* wK2T  = (unsigned short*)alloc((size_t)C * 32 * 2);
    unsigned short* wMa2T = (unsigned short*)alloc((size_t)C * 32 * 2);
    unsigned short* wMk2T = (unsigned short*)alloc((size_t)C * 32 * 2);

    // --- activations ---
    float*          dxf    = (float*)alloc((size_t)BT * C * 4);
    unsigned short* basebf = (unsigned short*)alloc((size_t)BT * C * 2); // reused as ybf
    float*          smallF = (float*)alloc((size_t)BT * 128 * 4);
    unsigned short* L1bf   = (unsigned short*)alloc((size_t)BT * 128 * 2);
    unsigned short* smallBF= (unsigned short*)alloc((size_t)BT * 128 * 2);
    float*          mtmp   = (float*)alloc((size_t)BT * C * 4);          // reused as y
    unsigned short* xrg    = (unsigned short*)alloc((size_t)BT * C * 2);
    unsigned short* xwa    = (unsigned short*)alloc((size_t)BT * C * 2);
    unsigned short* xk     = (unsigned short*)alloc((size_t)BT * C * 2);
    unsigned short* xv     = (unsigned short*)alloc((size_t)BT * C * 2);
    float*          rbuf   = (float*)alloc((size_t)BT * C * 4);
    float*          kbuf   = (float*)alloc((size_t)BT * C * 4);
    float*          vbuf   = (float*)alloc((size_t)BT * C * 4);
    float*          gbuf   = (float*)alloc((size_t)BT * C * 4);
    float*          d2buf  = (float*)alloc((size_t)BT * C * 4);          // -> wdec
    float*          a2buf  = (float*)alloc((size_t)BT * C * 4);
    float*          kk2buf = (float*)alloc((size_t)BT * C * 4);
    float*          ma2buf = (float*)alloc((size_t)BT * C * 4);
    float*          mk2buf = (float*)alloc((size_t)BT * C * 4);
    float*          kknbuf = (float*)alloc((size_t)BT * C * 4);
    float*          bbuf   = (float*)alloc((size_t)BT * C * 4);
    float*          ybuf   = mtmp;
    unsigned short* ybf    = basebf;

    const int TB = 256;
    auto blocks = [&](size_t n) { return (unsigned)((n + TB - 1) / TB); };

    // ---- weight conversion (N x K layout == original layout for W / w1) ----
    cvt_kernel<<<blocks((size_t)C * C), TB, 0, stream>>>(Wr, wWr, C * C);
    cvt_kernel<<<blocks((size_t)C * C), TB, 0, stream>>>(Wk, wWk, C * C);
    cvt_kernel<<<blocks((size_t)C * C), TB, 0, stream>>>(Wv, wWv, C * C);
    cvt_kernel<<<blocks((size_t)C * C), TB, 0, stream>>>(Wo, wWo, C * C);
    cvt_kernel<<<blocks((size_t)128 * C), TB, 0, stream>>>(maa_w1, wMaa1, 128 * C);
    cvt_kernel<<<blocks((size_t)4 * C * 32), TB, 0, stream>>>(maa_w2, wMaa2, 4 * C * 32);
    cvt_kernel<<<blocks((size_t)128 * C), TB, 0, stream>>>(gate_w1, wGate1, 128 * C);
    cvt_kernel<<<blocks((size_t)64 * C), TB, 0, stream>>>(dec_w1, wDec1, 64 * C);
    cvt_kernel<<<blocks((size_t)16 * C), TB, 0, stream>>>(aaa_w1, wA1, 16 * C);
    cvt_kernel<<<blocks((size_t)16 * C), TB, 0, stream>>>(kkk_w1, wK1, 16 * C);
    cvt_kernel<<<blocks((size_t)16 * C), TB, 0, stream>>>(ma_w1, wMa1, 16 * C);
    cvt_kernel<<<blocks((size_t)16 * C), TB, 0, stream>>>(mk_w1, wMk1, 16 * C);
    // w2-style weights are K x N -> transpose (and pad K where K=16)
    transpose_pad_kernel<<<blocks((size_t)C * 128), TB, 0, stream>>>(gate_w2, wGate2T, 128, 128, C);
    transpose_pad_kernel<<<blocks((size_t)C * 64), TB, 0, stream>>>(dec_w2, wDec2T, 64, 64, C);
    transpose_pad_kernel<<<blocks((size_t)C * 32), TB, 0, stream>>>(aaa_w2, wA2T, 16, 32, C);
    transpose_pad_kernel<<<blocks((size_t)C * 32), TB, 0, stream>>>(kkk_w2, wK2T, 16, 32, C);
    transpose_pad_kernel<<<blocks((size_t)C * 32), TB, 0, stream>>>(ma_w2, wMa2T, 16, 32, C);
    transpose_pad_kernel<<<blocks((size_t)C * 32), TB, 0, stream>>>(mk_w2, wMk2T, 16, 32, C);

    auto gemm = [&](const unsigned short* A, const unsigned short* B, float* Cp,
                    int M, int N, int K, int lda, int ldb, int ldc) {
        dim3 grid(M / BM, (N + BN - 1) / BN);
        gemm_bf16_kernel<<<grid, 256, 0, stream>>>(A, B, Cp, M, N, K, lda, ldb, ldc);
    };

    // ---- ddlerp ----
    prep_kernel<<<BT, 256, 0, stream>>>(x, time_maa_x, dxf, basebf, T, C);
    gemm(basebf, wMaa1, smallF, BT, 128, C, C, C, 128);
    act_cvt_kernel<<<blocks((size_t)BT * 128), TB, 0, stream>>>(smallF, L1bf, BT, 128, 128, 1);
    unsigned short* xmix[4] = { xrg, xwa, xk, xv };
    for (int i = 0; i < 4; i++) {
        gemm(L1bf + i * 32, wMaa2 + (size_t)i * C * 32, mtmp, BT, C, 32, 128, 32, C);
        mix_kernel<<<BT, 256, 0, stream>>>(x, dxf, time_maa + (size_t)i * C, mtmp, xmix[i], C);
    }

    // ---- main projections ----
    gemm(xrg, wWr, rbuf, BT, C, C, C, C, C);
    gemm(xk,  wWk, kbuf, BT, C, C, C, C, C);
    gemm(xv,  wWv, vbuf, BT, C, C, C, C, C);

    // gate: g = tanh(xrg @ gate_w1^T) @ gate_w2
    gemm(xrg, wGate1, smallF, BT, 128, C, C, C, 128);
    act_cvt_kernel<<<blocks((size_t)BT * 128), TB, 0, stream>>>(smallF, smallBF, BT, 128, 128, 1);
    gemm(smallBF, wGate2T, gbuf, BT, C, 128, 128, 128, C);

    // decay lora
    gemm(xwa, wDec1, smallF, BT, 64, C, C, C, 64);
    act_cvt_kernel<<<blocks((size_t)BT * 64), TB, 0, stream>>>(smallF, smallBF, BT, 64, 64, 1);
    gemm(smallBF, wDec2T, d2buf, BT, C, 64, 64, 64, C);

    // a lora (no tanh)
    gemm(xwa, wA1, smallF, BT, 16, C, C, C, 16);
    act_cvt_kernel<<<blocks((size_t)BT * 32), TB, 0, stream>>>(smallF, smallBF, BT, 16, 32, 0);
    gemm(smallBF, wA2T, a2buf, BT, C, 32, 32, 32, C);

    // kk lora (A = xk, no tanh)
    gemm(xk, wK1, smallF, BT, 16, C, C, C, 16);
    act_cvt_kernel<<<blocks((size_t)BT * 32), TB, 0, stream>>>(smallF, smallBF, BT, 16, 32, 0);
    gemm(smallBF, wK2T, kk2buf, BT, C, 32, 32, 32, C);

    // ma lora (A = xwa)
    gemm(xwa, wMa1, smallF, BT, 16, C, C, C, 16);
    act_cvt_kernel<<<blocks((size_t)BT * 32), TB, 0, stream>>>(smallF, smallBF, BT, 16, 32, 0);
    gemm(smallBF, wMa2T, ma2buf, BT, C, 32, 32, 32, C);

    // mk lora (A = xk)
    gemm(xk, wMk1, smallF, BT, 16, C, C, C, 16);
    act_cvt_kernel<<<blocks((size_t)BT * 32), TB, 0, stream>>>(smallF, smallBF, BT, 16, 32, 0);
    gemm(smallBF, wMk2T, mk2buf, BT, C, 32, 32, 32, C);

    // ---- expand / scan / norm ----
    expand_kernel<<<BT, 256, 0, stream>>>(time_decay, time_aaaaa, time_misc_a, time_misc_k,
                                          kbuf, kk2buf, d2buf, a2buf, ma2buf, mk2buf,
                                          kknbuf, bbuf, C);
    scan_kernel<<<Bb * H, 64, 0, stream>>>(d2buf, rbuf, kbuf, kknbuf, bbuf, vbuf,
                                           ybuf, Bb, T, H, C);
    gnorm_kernel<<<BT, 256, 0, stream>>>(ybuf, rbuf, kbuf, vbuf, gbuf, ln_w, ln_b,
                                         time_faaaa, ybf, C);

    // ---- final projection straight into d_out ----
    gemm(ybf, wWo, (float*)d_out, BT, C, C, C, C, C);
}
